// LinearSelfAttention_89885075570758
// MI455X (gfx1250) — matrix operand order repaired and checked by reading, not compiled, not run on hardware
//
#include <hip/hip_runtime.h>
#include <stdint.h>

#define BB 8
#define NN 4096
#define DD 256
#define SLAB 64

typedef uint32_t u32;
typedef uint16_t u16;

typedef __attribute__((ext_vector_type(16))) __bf16 bf16x16;
typedef __attribute__((ext_vector_type(8)))  __bf16 bf16x8;
typedef __attribute__((ext_vector_type(8)))  float  f32x8;
typedef __attribute__((ext_vector_type(4)))  float  f32x4;

union Frag {
  bf16x16 v;
  bf16x8  h[2];
  u32     d[8];
};

#if __has_builtin(__builtin_amdgcn_exp2f)
#define EXP2F(x) __builtin_amdgcn_exp2f(x)
#else
#define EXP2F(x) exp2f(x)
#endif

// --- bf16 pack via hardware cvt: 1 instruction / 2 elements ----------------
__device__ __forceinline__ u32 packbf(float a, float b) {
  u32 r;
  asm("v_cvt_pk_bf16_f32 %0, %1, %2" : "=v"(r) : "v"(a), "v"(b));
  return r;
}
__device__ __forceinline__ u16 f2bf(float x) { return (u16)packbf(x, 0.0f); }
__device__ __forceinline__ __bf16 f2bfh(float x) {
  u16 v = f2bf(x);
  return __builtin_bit_cast(__bf16, v);
}

__device__ __forceinline__ f32x8 wmma_bf16(bf16x16 a, bf16x16 b, f32x8 c) {
  return __builtin_amdgcn_wmma_f32_16x16x32_bf16(false, a, false, b, (short)0, c, false, false);
}

// Async DMA: 16 bytes global -> LDS per lane, tracked by ASYNCcnt.
__device__ __forceinline__ void async_ld_b128(unsigned lds_off, unsigned gbyte,
                                              unsigned long long sbase) {
  asm volatile("global_load_async_to_lds_b128 %0, %1, %2"
               :: "v"(lds_off), "v"(gbyte), "s"(sbase) : "memory");
}
__device__ __forceinline__ void wait_async0() {
  asm volatile("s_wait_asynccnt 0" ::: "memory");
}

// ---------------------------------------------------------------------------
// Projection: Out = bf16( X . W + bias ).  X:[32768,256] f32, W:[256,256] f32.
// transposed==0: Out[row][col] ([B*N][256] bf16)   (Ke)
// transposed==1: Out[b][col][n] ([B][256][N] bf16) (Vf, pre-transposed)
// ---------------------------------------------------------------------------
__global__ __launch_bounds__(256)
void proj_kernel(const float* __restrict__ X,
                 const float* __restrict__ W,
                 const float* __restrict__ bias,
                 u16* __restrict__ Out,
                 int transposed)
{
  __shared__ __bf16 Wt[256][264];   // W transposed: Wt[out][d]
  __shared__ float  biasLds[256];

  const int t = threadIdx.x;
  for (int q = 0; q < 64; ++q) {
    int lin = q * 256 + t;
    int d   = lin >> 6;
    int o4  = (lin & 63) << 2;
    f32x4 w4 = *(const f32x4*)(W + (size_t)d * 256 + o4);
    Wt[o4 + 0][d] = f2bfh(w4.x);
    Wt[o4 + 1][d] = f2bfh(w4.y);
    Wt[o4 + 2][d] = f2bfh(w4.z);
    Wt[o4 + 3][d] = f2bfh(w4.w);
  }
  biasLds[t] = bias[t];
  __syncthreads();

  const int wave = t >> 5;
  const int lane = t & 31;
  const int half = lane >> 4;
  const int l16  = lane & 15;
  const size_t rowbase = (size_t)blockIdx.x * 128 + wave * 16;
  const size_t row = rowbase + l16;

  f32x8 acc[16] = {};

  #pragma unroll
  for (int j = 0; j < 8; ++j) {
    const int db = 32 * j;
    const float* xp = X + row * 256 + db + (half ? 8 : 0);
    f32x4 a0 = *(const f32x4*)(xp + 0);
    f32x4 a1 = *(const f32x4*)(xp + 4);
    f32x4 a2 = *(const f32x4*)(xp + 16);
    f32x4 a3 = *(const f32x4*)(xp + 20);
    Frag af;
    af.d[0] = packbf(a0.x, a0.y); af.d[1] = packbf(a0.z, a0.w);
    af.d[2] = packbf(a1.x, a1.y); af.d[3] = packbf(a1.z, a1.w);
    af.d[4] = packbf(a2.x, a2.y); af.d[5] = packbf(a2.z, a2.w);
    af.d[6] = packbf(a3.x, a3.y); af.d[7] = packbf(a3.z, a3.w);

    Frag bw[2];
    {
      const __bf16* bp = &Wt[l16][db + (half ? 8 : 0)];
      bw[0].h[0] = *(const bf16x8*)bp; bw[0].h[1] = *(const bf16x8*)(bp + 16);
    }
    #pragma unroll
    for (int c = 0; c < 16; ++c) {
      if (c + 1 < 16) {
        const __bf16* bp = &Wt[(c + 1) * 16 + l16][db + (half ? 8 : 0)];
        bw[(c + 1) & 1].h[0] = *(const bf16x8*)bp;
        bw[(c + 1) & 1].h[1] = *(const bf16x8*)(bp + 16);
      }
      acc[c] = wmma_bf16(af.v, bw[c & 1].v, acc[c]);
    }
  }

  // branch-free epilogue: both layouts are affine in (m, col)
  size_t obase, rstr, cstr;
  if (!transposed) {
    obase = rowbase * 256; rstr = 256; cstr = 1;
  } else {
    size_t b  = rowbase >> 12;          // block never straddles a batch
    size_t n0 = rowbase & 4095;
    obase = (b * 256) * NN + n0; rstr = 1; cstr = NN;
  }
  #pragma unroll
  for (int c = 0; c < 16; ++c) {
    int col = c * 16 + l16;
    float bs = biasLds[col];
    #pragma unroll
    for (int r = 0; r < 8; ++r) {
      int m = r + half * 8;             // C layout: M = r + 8*(lane/16)
      Out[obase + (size_t)m * rstr + (size_t)col * cstr] = f2bf(acc[c][r] + bs);
    }
  }
}

// ---------------------------------------------------------------------------
// Flash attention: Out = softmax(Q.Ke^T/16) . Vf
// grid (32, 8): x = 128-row tile, y = batch. 8 waves, 16 rows/wave.
// Double-buffered 64-key slabs via GLOBAL_LOAD_ASYNC_TO_LDS_B128.
// Row sums of P computed by WMMA against an all-ones B fragment.
// ---------------------------------------------------------------------------
__global__ __launch_bounds__(256)
void attn_kernel(const float* __restrict__ Q,
                 const u16*  __restrict__ Ke,
                 const u16*  __restrict__ Vft,
                 float* __restrict__ Out)
{
  __shared__ __bf16 QLds[8][16][264];       // 67.6 KB
  __shared__ __bf16 KeLds[2][SLAB][264];    // 67.6 KB  [key][d]
  __shared__ __bf16 VtLds[2][256][72];      // 73.7 KB  [col][key]
  __shared__ __bf16 PLds[8][16][72];        // 18.4 KB

  const int t    = threadIdx.x;
  const int bidx = blockIdx.y;
  const int row0 = blockIdx.x * 128;
  const int wave = t >> 5;
  const int lane = t & 31;
  const int half = lane >> 4;
  const int l16  = lane & 15;

  const float SCALE = 0.0625f * 1.44269504088896340736f;  // (1/sqrt(256))*log2(e)

  auto issue_slab = [&](int buf, int kb) {
    unsigned long long keB =
        (unsigned long long)(size_t)(Ke + ((size_t)bidx * NN + kb) * DD);
    unsigned keL = (unsigned)(size_t)&KeLds[buf][0][0];
    #pragma unroll
    for (int j = 0; j < 8; ++j) {
      int lin = j * 256 + t;            // 16B chunk id
      int row = lin >> 5;               // key 0..63
      int c8  = (lin & 31) << 3;        // d
      async_ld_b128(keL + (unsigned)(row * (264 * 2) + c8 * 2),
                    (unsigned)((row * DD + c8) * 2), keB);
    }
    unsigned long long vtB =
        (unsigned long long)(size_t)(Vft + (size_t)bidx * 256 * NN + kb);
    unsigned vtL = (unsigned)(size_t)&VtLds[buf][0][0];
    #pragma unroll
    for (int j = 0; j < 8; ++j) {
      int lin = j * 256 + t;
      int col = lin >> 3;               // 0..255
      int k8  = (lin & 7) << 3;         // key
      async_ld_b128(vtL + (unsigned)(col * (72 * 2) + k8 * 2),
                    (unsigned)((col * NN + k8) * 2), vtB);
    }
  };

  issue_slab(0, 0);   // slab 0 in flight while we stage Q

  // ---- stage Q tile (128 x 256) as scaled bf16 ----
  #pragma unroll 4
  for (int it = 0; it < 32; ++it) {
    int lin = it * 256 + t;
    int row = lin >> 6;
    int c4  = (lin & 63) << 2;
    f32x4 q = *(const f32x4*)(Q + ((size_t)bidx * NN + row0 + row) * DD + c4);
    int w = row >> 4, rr = row & 15;
    *(u32*)&QLds[w][rr][c4]     = packbf(q.x * SCALE, q.y * SCALE);
    *(u32*)&QLds[w][rr][c4 + 2] = packbf(q.z * SCALE, q.w * SCALE);
  }

  Frag ones;                       // all-ones B fragment for row-sum WMMA
  #pragma unroll
  for (int j = 0; j < 8; ++j) ones.d[j] = 0x3F803F80u;

  float m_r[8], l_r[8];
  #pragma unroll
  for (int r = 0; r < 8; ++r) { m_r[r] = -__builtin_inff(); l_r[r] = 0.0f; }
  f32x8 acc[16] = {};

  const int NIT = NN / SLAB;   // 64
  for (int i = 0; i < NIT; ++i) {
    const int cur = i & 1;
    wait_async0();             // my slab-i chunks are in LDS
    __syncthreads();           // everyone's slab-i data visible; other buf free
    if (i + 1 < NIT) issue_slab(cur ^ 1, (i + 1) * SLAB);

    const __bf16 (*KeT)[264] = KeLds[cur];
    const __bf16 (*VtT)[72]  = VtLds[cur];

    // ---- S = Qtile . Ke^T  (16 x 64) ----
    f32x8 s[4] = {};
    #pragma unroll
    for (int j = 0; j < 8; ++j) {
      const __bf16* qp = &QLds[wave][l16][32 * j + (half ? 8 : 0)];
      Frag aq; aq.h[0] = *(const bf16x8*)qp; aq.h[1] = *(const bf16x8*)(qp + 16);
      Frag bk[4];
      #pragma unroll
      for (int ct = 0; ct < 4; ++ct) {
        const __bf16* kp = &KeT[ct * 16 + l16][32 * j + (half ? 8 : 0)];
        bk[ct].h[0] = *(const bf16x8*)kp; bk[ct].h[1] = *(const bf16x8*)(kp + 16);
      }
      #pragma unroll
      for (int ct = 0; ct < 4; ++ct) s[ct] = wmma_bf16(aq.v, bk[ct].v, s[ct]);
    }

    // ---- online softmax: row max via lane shuffles, exp in base-2 ----
    float p[4][8], alpha[8];
    #pragma unroll
    for (int r = 0; r < 8; ++r) {
      float tmax = fmaxf(fmaxf(s[0][r], s[1][r]), fmaxf(s[2][r], s[3][r]));
      #pragma unroll
      for (int off = 1; off < 16; off <<= 1)
        tmax = fmaxf(tmax, __shfl_xor(tmax, off, 32));
      float nm = fmaxf(m_r[r], tmax);
      alpha[r] = EXP2F(m_r[r] - nm);
      m_r[r]   = nm;
      #pragma unroll
      for (int ct = 0; ct < 4; ++ct) p[ct][r] = EXP2F(s[ct][r] - nm);
    }

    // ---- stash P (C layout -> A layout via LDS) ----
    #pragma unroll
    for (int r = 0; r < 8; ++r) {
      int m = r + half * 8;
      #pragma unroll
      for (int ct = 0; ct < 4; ++ct)
        PLds[wave][m][ct * 16 + l16] = f2bfh(p[ct][r]);
    }

    // ---- rescale acc only when the running max actually moved (uniform) ----
    int needscale = 0;
    #pragma unroll
    for (int r = 0; r < 8; ++r) needscale |= (alpha[r] != 1.0f);
    if (__any(needscale)) {
      #pragma unroll
      for (int c = 0; c < 16; ++c) {
        #pragma unroll
        for (int r = 0; r < 8; ++r) acc[c][r] *= alpha[r];
      }
    }
    asm volatile("s_wait_dscnt 0" ::: "memory");

    Frag ap0, ap1;
    {
      const __bf16* pp = &PLds[wave][l16][half ? 16 : 0];
      ap0.h[0] = *(const bf16x8*)pp;        ap0.h[1] = *(const bf16x8*)(pp + 8);
      ap1.h[0] = *(const bf16x8*)(pp + 32); ap1.h[1] = *(const bf16x8*)(pp + 40);
    }

    // ---- row sums of quantized P via WMMA (replicated across lanes) ----
    f32x8 rsum = {};
    rsum = wmma_bf16(ap0.v, ones.v, rsum);
    rsum = wmma_bf16(ap1.v, ones.v, rsum);
    #pragma unroll
    for (int r = 0; r < 8; ++r) l_r[r] = l_r[r] * alpha[r] + rsum[r];

    // ---- acc += P . Vf : ping-pong B-fragment pairs ----
    Frag bv[4];
    {
      const __bf16* vp = &VtT[l16][half ? 16 : 0];
      bv[0].h[0] = *(const bf16x8*)vp;        bv[0].h[1] = *(const bf16x8*)(vp + 8);
      bv[1].h[0] = *(const bf16x8*)(vp + 32); bv[1].h[1] = *(const bf16x8*)(vp + 40);
    }
    #pragma unroll
    for (int c = 0; c < 16; ++c) {
      int cs = (c & 1) * 2;
      if (c + 1 < 16) {
        const __bf16* vp = &VtT[(c + 1) * 16 + l16][half ? 16 : 0];
        int ns = ((c + 1) & 1) * 2;
        bv[ns + 0].h[0] = *(const bf16x8*)vp;        bv[ns + 0].h[1] = *(const bf16x8*)(vp + 8);
        bv[ns + 1].h[0] = *(const bf16x8*)(vp + 32); bv[ns + 1].h[1] = *(const bf16x8*)(vp + 40);
      }
      acc[c] = wmma_bf16(ap0.v, bv[cs + 0].v, acc[c]);
      acc[c] = wmma_bf16(ap1.v, bv[cs + 1].v, acc[c]);
    }
  }

  // ---- normalize rows and store f32 output ----
  float inv[8];
  #pragma unroll
  for (int r = 0; r < 8; ++r) inv[r] = 1.0f / l_r[r];
  float* op = Out + ((size_t)bidx * NN + row0 + wave * 16) * DD;
  #pragma unroll
  for (int c = 0; c < 16; ++c) {
    int col = c * 16 + l16;
    #pragma unroll
    for (int r = 0; r < 8; ++r) {
      int m = r + half * 8;
      op[(size_t)m * DD + col] = acc[c][r] * inv[r];
    }
  }
}

// ---------------------------------------------------------------------------
extern "C" void kernel_launch(void* const* d_in, const int* in_sizes, int n_in,
                              void* d_out, int out_size, void* d_ws, size_t ws_size,
                              hipStream_t stream) {
  const float* K  = (const float*)d_in[0];
  const float* Q  = (const float*)d_in[1];
  const float* V  = (const float*)d_in[2];
  const float* We = (const float*)d_in[3];
  const float* be = (const float*)d_in[4];
  const float* Wf = (const float*)d_in[5];
  const float* bf = (const float*)d_in[6];
  float* out = (float*)d_out;

  u16* KeW  = (u16*)d_ws;                              // [B*N][256] bf16
  u16* VftW = KeW + (size_t)BB * NN * DD;              // [B][256][N] bf16

  proj_kernel<<<dim3(256, 1, 1), 256, 0, stream>>>(K, We, be, KeW, 0);
  proj_kernel<<<dim3(256, 1, 1), 256, 0, stream>>>(V, Wf, bf, VftW, 1);
  attn_kernel<<<dim3(32, 8, 1), 256, 0, stream>>>(Q, KeW, VftW, out);
}
